// Mamba_16149077033106
// MI455X (gfx1250) — matrix operand order repaired and checked by reading, not compiled, not run on hardware
//
#include <hip/hip_runtime.h>
#include <hip/hip_bf16.h>
#include <math.h>

// ---------------- Model dims (compile-time) ----------------
#define L_SEQ    1024
#define D_MODEL  1024
#define D_STATE  16
#define D_CONV   4
#define D_INNER  2048          // 2*D_MODEL
#define DT_RANK  64
#define XPROJ_N  (DT_RANK + 2 * D_STATE)   // 96
#define T_CHUNK  16            // scan: timesteps staged per barrier round

// ---------------- Vector types ----------------
typedef __attribute__((ext_vector_type(16))) __bf16        v16bf;
typedef __attribute__((ext_vector_type(8)))  float         v8f;
typedef __attribute__((ext_vector_type(2)))  float         f32x2;
typedef __attribute__((ext_vector_type(4)))  unsigned int  u32x4;
typedef __attribute__((ext_vector_type(8)))  unsigned int  u32x8;

// ============================================================
// Tensor Data Mover: 2D tile load (global fp32 -> LDS), D# per
// CDNA5 ISA.  data_size = 4 bytes.
// OOB rows/cols (>= tensor_d*) read as zero -> free zero-padding.
// Issued once per block (TDM ignores EXEC); tracked by TENSORcnt.
// ============================================================
__device__ __forceinline__ void tdm_load_2d(unsigned lds_off, const void* gptr,
                                            unsigned tile_d0, unsigned tile_d1,
                                            unsigned tensor_d0, unsigned tensor_d1,
                                            unsigned long long stride_elems)
{
    unsigned long long ga = (unsigned long long)gptr;
    u32x4 g0;
    g0[0] = 1u;                                        // count=1 (valid), user mode
    g0[1] = lds_off;                                   // lds_addr (bytes)
    g0[2] = (unsigned)(ga & 0xFFFFFFFFu);              // global_addr[31:0]
    g0[3] = (unsigned)((ga >> 32) & 0x01FFFFFFu)       // global_addr[56:32]
          | 0x80000000u;                               // type = 2 ("image")

    unsigned long long q0 = (2ull << 16)                                   // data_size=4B
                          | ((unsigned long long)(tensor_d0 & 0xFFFFu) << 48);
    unsigned long long q1 = ((unsigned long long)(tensor_d0 >> 16) & 0xFFFFull)
                          | ((unsigned long long)tensor_d1 << 16)          // tensor_dim1
                          | ((unsigned long long)(tile_d0 & 0xFFFFu) << 48);
    unsigned long long q2 = (unsigned long long)(tile_d1 & 0xFFFFu)        // tile_dim1
                          | ((stride_elems & 0xFFFFFFFFull) << 32);        // dim0_stride lo
    unsigned long long q3 = (stride_elems >> 32) & 0xFFFFull;              // dim0_stride hi
    u32x8 g1;
    g1[0] = (unsigned)q0; g1[1] = (unsigned)(q0 >> 32);
    g1[2] = (unsigned)q1; g1[3] = (unsigned)(q1 >> 32);
    g1[4] = (unsigned)q2; g1[5] = (unsigned)(q2 >> 32);
    g1[6] = (unsigned)q3; g1[7] = (unsigned)(q3 >> 32);

    asm volatile("tensor_load_to_lds %0, %1" :: "s"(g0), "s"(g1) : "memory");
}

// ============================================================
// bf16-WMMA GEMM:  C[1024 x ND] = A[1024 x KD] * W[ND x KD]^T
//   Double-buffered TDM pipeline: while WMMA consumes buffer b,
//   the TDM streams K-slab k0+32 into buffer b^1; the in-order
//   TENSORcnt lets us wait "<= 2" to release exactly buffer b.
//   Block tile 64 x 128, K-step 32; 8 wave32s, each 32x32 (4 WMMA).
//   EPI == 1: fused softplus(x + bias[n]).
// ============================================================
template <int EPI, int LDA, int KD, int ND>
__global__ __launch_bounds__(256) void wmma_gemm_bf16(
    const float* __restrict__ A,
    const float* __restrict__ W,
    const float* __restrict__ bias,
    float* __restrict__ C)
{
    __shared__ float sA[2][64 * 32];     // [buf][m][k]  2 x  8 KB
    __shared__ float sB[2][128 * 32];    // [buf][n][k]  2 x 16 KB

    const int tid  = threadIdx.x;
    const int m0   = blockIdx.x * 64;
    const int n0   = blockIdx.y * 128;
    const int wave = tid >> 5;
    const int lane = tid & 31;
    const int wm   = wave >> 2;       // 0..1
    const int wn   = wave & 3;        // 0..3
    const int rl   = lane & 15;
    const int hs   = lane >> 4;       // lanes 0-15 vs 16-31

    const unsigned ldsA0 = (unsigned)(unsigned long long)(const void*)&sA[0][0];
    const unsigned ldsA1 = (unsigned)(unsigned long long)(const void*)&sA[1][0];
    const unsigned ldsB0 = (unsigned)(unsigned long long)(const void*)&sB[0][0];
    const unsigned ldsB1 = (unsigned)(unsigned long long)(const void*)&sB[1][0];

    v8f acc[2][2] = {};

    // ---- prologue: fill buffer 0 with first K-slab ----
    if (tid == 0) {
        tdm_load_2d(ldsA0, &A[(size_t)m0 * LDA], 32, 64,
                    (unsigned)KD, 1024u, (unsigned long long)LDA);
        tdm_load_2d(ldsB0, &W[(size_t)n0 * KD], 32, 128,
                    (unsigned)KD, (unsigned)ND, (unsigned long long)KD);
    }

    int buf = 0;
    for (int k0 = 0; k0 < KD; k0 += 32, buf ^= 1) {
        const bool has_next = (k0 + 32 < KD);
        // ---- issue next slab into the other buffer (overlaps compute) ----
        if (has_next && tid == 0) {
            tdm_load_2d(buf ? ldsA0 : ldsA1, &A[(size_t)m0 * LDA + (k0 + 32)],
                        32, 64, (unsigned)KD, 1024u, (unsigned long long)LDA);
            tdm_load_2d(buf ? ldsB0 : ldsB1, &W[(size_t)n0 * KD + (k0 + 32)],
                        32, 128, (unsigned)KD, (unsigned)ND,
                        (unsigned long long)KD);
        }
        // ---- release current buffer: its 2 TDM ops are the oldest ----
        if (wave == 0) {
            if (has_next) __builtin_amdgcn_s_wait_tensorcnt(2);
            else          __builtin_amdgcn_s_wait_tensorcnt(0);
        }
        __syncthreads();

        const float* __restrict__ tA = &sA[buf][0];
        const float* __restrict__ tB = &sB[buf][0];

        // ---- fragments: ds_load fp32 k-pairs, pack to bf16 ----
        v16bf af[2], bfr[2];
        #pragma unroll
        for (int f = 0; f < 2; ++f) {
            const int arow = wm * 32 + f * 16 + rl;
            #pragma unroll
            for (int v = 0; v < 8; ++v) {
                // A 16x32 bf16 fragment layout (ISA 7.12.2)
                int kk = ((v < 4) ? 0 : 16) + (hs ? 8 : 0) + 2 * (v & 3);
                f32x2 p = *(const f32x2*)&tA[arow * 32 + kk];
                af[f][2 * v]     = (__bf16)p[0];
                af[f][2 * v + 1] = (__bf16)p[1];
            }
            const int bcol = wn * 32 + f * 16 + rl;
            #pragma unroll
            for (int v = 0; v < 8; ++v) {
                // B 32x16 layout: same K order as A fragment (ISA 7.12.2)
                int kk = ((v < 4) ? 0 : 16) + (hs ? 8 : 0) + 2 * (v & 3);
                f32x2 p = *(const f32x2*)&tB[bcol * 32 + kk];
                bfr[f][2 * v]     = (__bf16)p[0];
                bfr[f][2 * v + 1] = (__bf16)p[1];
            }
        }

        #pragma unroll
        for (int i = 0; i < 2; ++i)
            #pragma unroll
            for (int j = 0; j < 2; ++j)
                acc[i][j] = __builtin_amdgcn_wmma_f32_16x16x32_bf16(
                    false, af[i], false, bfr[j], (short)0, acc[i][j],
                    false, false);
        __syncthreads();   // all waves done with buf before it is refilled
    }

    // ---- epilogue + store: D layout M = r + 8*(lane/16), N = lane%16 ----
    const int mloc = hs * 8;
    #pragma unroll
    for (int i = 0; i < 2; ++i)
        #pragma unroll
        for (int j = 0; j < 2; ++j) {
            const int gn = n0 + wn * 32 + j * 16 + rl;
            if (gn < ND) {
                #pragma unroll
                for (int r = 0; r < 8; ++r) {
                    const int gm = m0 + wm * 32 + i * 16 + mloc + r;
                    float v = acc[i][j][r];
                    if (EPI == 1) {
                        v += bias[gn];
                        v = (v > 20.0f) ? v : log1pf(__expf(v));  // softplus
                    }
                    C[(size_t)gm * ND + gn] = v;
                }
            }
        }
}

// ============================================================
// Causal depthwise conv1d (width 4) + SiLU on the x-branch of xz.
// ============================================================
__global__ __launch_bounds__(256) void conv_silu_kernel(
    const float* __restrict__ xz,
    const float* __restrict__ conv_w,   // [D_INNER, 4]
    const float* __restrict__ conv_b,   // [D_INNER]
    float* __restrict__ xi)
{
    const int idx = blockIdx.x * 256 + threadIdx.x;
    const int l = idx / D_INNER;
    const int d = idx % D_INNER;
    float acc = conv_b[d];
    #pragma unroll
    for (int j = 0; j < D_CONV; ++j) {
        int ll = l - (D_CONV - 1) + j;
        if (ll >= 0)
            acc += conv_w[d * D_CONV + j] * xz[(size_t)ll * (2 * D_INNER) + d];
    }
    xi[(size_t)l * D_INNER + d] = acc / (1.0f + __expf(-acc));   // SiLU
}

// ============================================================
// Selective scan, fused D-skip + SiLU gate.  Sequential over t,
// parallel over 2048 channels.  B_t/C_t staged in LDS in chunks
// of T_CHUNK timesteps -> 2 barriers per 16 steps instead of per step.
// ============================================================
__global__ __launch_bounds__(256) void scan_kernel(
    const float* __restrict__ delta,    // [L, D_INNER]
    const float* __restrict__ xi,       // [L, D_INNER]
    const float* __restrict__ x_dbl,    // [L, 96]  (dt | B | C)
    const float* __restrict__ xz,       // [L, 2*D_INNER]  (gate z)
    const float* __restrict__ A_log,    // [D_INNER, 16]
    const float* __restrict__ D_skip,   // [D_INNER]
    float* __restrict__ yg)             // [L, D_INNER]
{
    const int d = blockIdx.x * 256 + threadIdx.x;
    __shared__ float sBC[T_CHUNK * 2 * D_STATE];       // [t][B(16)|C(16)]

    float Ad[D_STATE];
    #pragma unroll
    for (int n = 0; n < D_STATE; ++n)
        Ad[n] = -__expf(A_log[(size_t)d * D_STATE + n]);

    float h[D_STATE];
    #pragma unroll
    for (int n = 0; n < D_STATE; ++n) h[n] = 0.0f;

    const float dsk = D_skip[d];

    for (int tb = 0; tb < L_SEQ; tb += T_CHUNK) {
        // stage B/C for T_CHUNK timesteps (512 floats, 2 per thread)
        #pragma unroll
        for (int e = threadIdx.x; e < T_CHUNK * 2 * D_STATE; e += 256) {
            const int tt = e >> 5, c = e & 31;
            sBC[e] = x_dbl[(size_t)(tb + tt) * XPROJ_N + DT_RANK + c];
        }
        __syncthreads();

        for (int t = tb; t < tb + T_CHUNK; ++t) {
            const float* bc = &sBC[(t - tb) * 2 * D_STATE];
            const float dl = delta[(size_t)t * D_INNER + d];
            const float xv = xi[(size_t)t * D_INNER + d];
            const float dx = dl * xv;
            float y = 0.0f;
            #pragma unroll
            for (int n = 0; n < D_STATE; ++n) {
                const float dA = __expf(dl * Ad[n]);
                h[n] = dA * h[n] + dx * bc[n];
                y = fmaf(h[n], bc[D_STATE + n], y);
            }
            y = fmaf(xv, dsk, y);
            const float z = xz[(size_t)t * (2 * D_INNER) + D_INNER + d];
            yg[(size_t)t * D_INNER + d] = y * (z / (1.0f + __expf(-z)));
        }
        __syncthreads();
    }
}

// ============================================================
// Host-side launch: 6-stage pipeline on `stream`.
// Workspace (floats): xz 4,194,304 | xi 2,097,152 | x_dbl 98,304
//                     delta 2,097,152 | yg 2,097,152  (~42.3 MB)
// ============================================================
extern "C" void kernel_launch(void* const* d_in, const int* in_sizes, int n_in,
                              void* d_out, int out_size, void* d_ws, size_t ws_size,
                              hipStream_t stream)
{
    const float* x      = (const float*)d_in[0];
    const float* W_in   = (const float*)d_in[1];
    const float* conv_w = (const float*)d_in[2];
    const float* conv_b = (const float*)d_in[3];
    const float* W_xprj = (const float*)d_in[4];
    const float* W_dt   = (const float*)d_in[5];
    const float* b_dt   = (const float*)d_in[6];
    const float* A_log  = (const float*)d_in[7];
    const float* D_skip = (const float*)d_in[8];
    const float* W_out  = (const float*)d_in[9];
    float* out = (float*)d_out;

    float* ws    = (float*)d_ws;
    float* xz    = ws;
    float* xi    = xz    + (size_t)L_SEQ * (2 * D_INNER);
    float* xdbl  = xi    + (size_t)L_SEQ * D_INNER;
    float* delta = xdbl  + (size_t)L_SEQ * XPROJ_N;
    float* yg    = delta + (size_t)L_SEQ * D_INNER;

    const dim3 blk(256);

    // 1) xz = x @ W_in^T                       (M=1024, N=4096, K=1024)
    wmma_gemm_bf16<0, D_MODEL, D_MODEL, 2 * D_INNER>
        <<<dim3(L_SEQ / 64, (2 * D_INNER) / 128), blk, 0, stream>>>(
            x, W_in, nullptr, xz);

    // 2) causal depthwise conv + SiLU -> xi
    conv_silu_kernel<<<(L_SEQ * D_INNER) / 256, blk, 0, stream>>>(
        xz, conv_w, conv_b, xi);

    // 3) x_dbl = xi @ W_xproj^T                (N=96, K=2048)
    wmma_gemm_bf16<0, D_INNER, D_INNER, XPROJ_N>
        <<<dim3(L_SEQ / 64, 1), blk, 0, stream>>>(
            xi, W_xprj, nullptr, xdbl);

    // 4) delta = softplus(dt @ W_dt^T + b_dt)  (N=2048, K=64, A = x_dbl[:, :64])
    wmma_gemm_bf16<1, XPROJ_N, DT_RANK, D_INNER>
        <<<dim3(L_SEQ / 64, D_INNER / 128), blk, 0, stream>>>(
            xdbl, W_dt, b_dt, delta);

    // 5) selective scan + D-skip + SiLU gate -> yg
    scan_kernel<<<D_INNER / 256, blk, 0, stream>>>(
        delta, xi, xdbl, xz, A_log, D_skip, yg);

    // 6) out = yg @ W_out^T                    (N=1024, K=2048)
    wmma_gemm_bf16<0, D_INNER, D_INNER, D_MODEL>
        <<<dim3(L_SEQ / 64, D_MODEL / 128), blk, 0, stream>>>(
            yg, W_out, nullptr, out);
}
